// GatClassifier_32573031973516
// MI455X (gfx1250) — matrix operand order repaired and checked by reading, not compiled, not run on hardware
//
#include <hip/hip_runtime.h>
#include <string.h>

// ---------------------------------------------------------------------------
// GAT classifier for gfx1250 (MI455X).
//   * GEMMs in bf16 via v_wmma_f32_16x16x32_bf16 (f32 accumulate),
//     register-blocked: one wave owns a 32x64 output tile (2x4 WMMA tiles),
//     so each K-step issues 8 WMMAs on 12 b128 fragment loads.
//   * segment softmax via ordered-uint atomicMax + f32 atomicAdd
//   * edge aggregation: one wave32 per edge, f32 atomics into agg
//   * mish+LayerNorm epilogues: one wave per node, shfl_xor reductions
// ---------------------------------------------------------------------------

#define N_HIDDEN   128
#define HEADS_12   4
#define HH         (HEADS_12 * N_HIDDEN)   // 512
#define OUT_DIM    16
#define NEG_SLOPE  0.2f
#define EPS_LN     1e-5f
#define EPS_SM     1e-16f

#define GEMM_MR    2    // 16-row subtiles per wave  (32 rows)
#define GEMM_NC    4    // 16-col subtiles per wave  (64 cols)

typedef __attribute__((ext_vector_type(16))) __bf16 v16bf;
typedef __attribute__((ext_vector_type(8)))  float  v8f;

union FragAB { v16bf v; uint4 q[2]; };
union FragC  { v8f   v; float f[8]; };
union BfBits { unsigned short u; __bf16 b; };

__device__ __forceinline__ __bf16 f2bf(float f) {
    unsigned u = __float_as_uint(f);
    u += 0x7FFFu + ((u >> 16) & 1u);          // round to nearest even
    BfBits x; x.u = (unsigned short)(u >> 16); return x.b;
}
__device__ __forceinline__ float bf2f(__bf16 b) {
    BfBits x; x.b = b; return __uint_as_float(((unsigned)x.u) << 16);
}
// order-preserving float <-> uint for atomicMax-based segment max
__device__ __forceinline__ unsigned encOrd(float f) {
    unsigned u = __float_as_uint(f);
    return (u & 0x80000000u) ? ~u : (u | 0x80000000u);
}
__device__ __forceinline__ float decOrd(unsigned u) {
    unsigned b = (u & 0x80000000u) ? (u ^ 0x80000000u) : ~u;
    return __uint_as_float(b);
}
__device__ __forceinline__ float mishf(float x) {
    float sp = (x > 20.f) ? x : log1pf(__expf(x));
    return x * tanhf(sp);
}
__device__ __forceinline__ float lrelu(float x) { return x < 0.f ? x * NEG_SLOPE : x; }

// ---------------------------------------------------------------------------
// f32 -> bf16 convert (grid-stride)
__global__ void cvt_bf16_kernel(const float* __restrict__ x, __bf16* __restrict__ y, long n) {
    long i = (long)blockIdx.x * blockDim.x + threadIdx.x;
    long st = (long)gridDim.x * blockDim.x;
    for (; i < n; i += st) y[i] = f2bf(x[i]);
}
// W[K][M] f32 -> Wt[M][K] bf16
__global__ void transpose_bf16_kernel(const float* __restrict__ W, __bf16* __restrict__ Wt,
                                      int K, int M) {
    int i = blockIdx.x * blockDim.x + threadIdx.x;
    if (i >= K * M) return;
    int k = i / M, m = i % M;
    Wt[(size_t)m * K + k] = f2bf(W[i]);
}
__global__ void zero_f32_kernel(float* __restrict__ p, long n) {
    long i = (long)blockIdx.x * blockDim.x + threadIdx.x;
    long st = (long)gridDim.x * blockDim.x;
    for (; i < n; i += st) p[i] = 0.f;
}

// ---------------------------------------------------------------------------
// C[N,M](bf16) = A[N,K](bf16) @ Bt[M,K](bf16)^T
// One wave computes a 32x64 output tile: acc[2][4] of 16x16 WMMA tiles.
// Per K-step (32): 4 A-fragment loads + 8 B-fragment loads feed 8 WMMAs.
__global__ __launch_bounds__(256)
void gemm_wmma_bf16_kernel(const __bf16* __restrict__ A, const __bf16* __restrict__ Bt,
                           __bf16* __restrict__ C, int N, int K, int M) {
    const int lane    = threadIdx.x & 31;
    const int wave    = threadIdx.x >> 5;
    const int wTilesM = M >> 6;                     // 64 output cols per wave
    const int wid     = blockIdx.x * 8 + wave;
    const int rowW    = wid / wTilesM;
    const int colW    = wid % wTilesM;
    const int R       = rowW * (GEMM_MR * 16);
    if (R >= N) return;                             // wave-uniform exit

    const int half = lane >> 4;                     // 0 | 1
    const int lm   = lane & 15;

    const __bf16* Arow[GEMM_MR];
    const __bf16* Brow[GEMM_NC];
    #pragma unroll
    for (int mt = 0; mt < GEMM_MR; ++mt) {
        int r = min(R + mt * 16 + lm, N - 1);       // clamp tail rows (load only)
        Arow[mt] = A + (size_t)r * K;
    }
    #pragma unroll
    for (int nt = 0; nt < GEMM_NC; ++nt) {
        int c = colW * 64 + nt * 16 + lm;
        Brow[nt] = Bt + (size_t)c * K + half * 16;
    }

    v8f acc[GEMM_MR][GEMM_NC];
    #pragma unroll
    for (int mt = 0; mt < GEMM_MR; ++mt)
        #pragma unroll
        for (int nt = 0; nt < GEMM_NC; ++nt)
            acc[mt][nt] = (v8f){};

    for (int k = 0; k < K; k += 32) {
        FragAB a[GEMM_MR], b[GEMM_NC];
        #pragma unroll
        for (int mt = 0; mt < GEMM_MR; ++mt) {
            // A 16x32 bf16 frag: half 0 -> K k..k+15; half 1 -> k+16..k+31
            a[mt].q[0] = *(const uint4*)(Arow[mt] + k + half * 16);
            a[mt].q[1] = *(const uint4*)(Arow[mt] + k + half * 16 + 8);
        }
        #pragma unroll
        for (int nt = 0; nt < GEMM_NC; ++nt) {
            // B 32x16 bf16 frag: half 0 -> K k..k+15; half 1 -> k+16..k+31
            b[nt].q[0] = *(const uint4*)(Brow[nt] + k);
            b[nt].q[1] = *(const uint4*)(Brow[nt] + k + 8);
        }
        #pragma unroll
        for (int mt = 0; mt < GEMM_MR; ++mt)
            #pragma unroll
            for (int nt = 0; nt < GEMM_NC; ++nt)
                acc[mt][nt] = __builtin_amdgcn_wmma_f32_16x16x32_bf16(
                    false, a[mt].v, false, b[nt].v, (short)0, acc[mt][nt], false, false);
    }

    // C 16x16 f32 layout per tile: VGPR r -> row r + half*8, col = lm
    #pragma unroll
    for (int mt = 0; mt < GEMM_MR; ++mt) {
        const int rbase = R + mt * 16 + half * 8;
        #pragma unroll
        for (int nt = 0; nt < GEMM_NC; ++nt) {
            FragC c; c.v = acc[mt][nt];
            const int col = colW * 64 + nt * 16 + lm;
            #pragma unroll
            for (int r = 0; r < 8; ++r) {
                int rr = rbase + r;
                if (rr < N) C[(size_t)rr * M + col] = f2bf(c.f[r]);
            }
        }
    }
}

// ---------------------------------------------------------------------------
// per (node, head): attention logits s,d; init segment-max / denom
__global__ void sd_init_kernel(const __bf16* __restrict__ h,
                               const float* __restrict__ a_src, const float* __restrict__ a_dst,
                               float* __restrict__ s, float* __restrict__ d,
                               unsigned* __restrict__ mEnc, float* __restrict__ den,
                               int N, int H) {
    int i = blockIdx.x * blockDim.x + threadIdx.x;
    if (i >= N * H) return;
    int n = i / H, hd = i % H;
    const __bf16* hp = h + (size_t)n * H * N_HIDDEN + hd * N_HIDDEN;
    const float* as = a_src + hd * N_HIDDEN;
    const float* ad = a_dst + hd * N_HIDDEN;
    float ss = 0.f, dd = 0.f;
    for (int f = 0; f < N_HIDDEN; ++f) {
        float v = bf2f(hp[f]);
        ss += v * as[f]; dd += v * ad[f];
    }
    s[i] = ss; d[i] = dd;
    mEnc[i] = encOrd(-__builtin_huge_valf());
    den[i] = 0.f;
}

// pass 1: segment max over dst (ordered-uint atomicMax)
__global__ void edge_max_kernel(const int* __restrict__ eidx,
                                const float* __restrict__ s, const float* __restrict__ d,
                                unsigned* __restrict__ mEnc, int E0, int Etot, int H) {
    int e = blockIdx.x * blockDim.x + threadIdx.x;
    if (e >= Etot) return;
    int src, dst;
    if (e < E0) { src = eidx[e]; dst = eidx[E0 + e]; } else { src = dst = e - E0; }
    for (int h = 0; h < H; ++h) {
        float v = lrelu(s[src * H + h] + d[dst * H + h]);
        atomicMax(&mEnc[dst * H + h], encOrd(v));
    }
}
// pass 2: segment sum of exp(e - m)
__global__ void edge_denom_kernel(const int* __restrict__ eidx,
                                  const float* __restrict__ s, const float* __restrict__ d,
                                  const unsigned* __restrict__ mEnc, float* __restrict__ den,
                                  int E0, int Etot, int H) {
    int e = blockIdx.x * blockDim.x + threadIdx.x;
    if (e >= Etot) return;
    int src, dst;
    if (e < E0) { src = eidx[e]; dst = eidx[E0 + e]; } else { src = dst = e - E0; }
    for (int h = 0; h < H; ++h) {
        float v = lrelu(s[src * H + h] + d[dst * H + h]);
        float m = decOrd(mEnc[dst * H + h]);
        atomicAdd(&den[dst * H + h], __expf(v - m));
    }
}
// pass 3: agg[dst] += alpha * h[src]   — one wave per edge, CH/32 feats per lane
__global__ __launch_bounds__(256)
void aggregate_kernel(const int* __restrict__ eidx, const __bf16* __restrict__ h,
                      const float* __restrict__ s, const float* __restrict__ d,
                      const unsigned* __restrict__ mEnc, const float* __restrict__ den,
                      float* __restrict__ agg, int E0, int Etot, int H, int CH) {
    int gw   = (int)(((long)blockIdx.x * blockDim.x + threadIdx.x) >> 5);
    int lane = threadIdx.x & 31;
    if (gw >= Etot) return;
    int src, dst;
    if (gw < E0) { src = eidx[gw]; dst = eidx[E0 + gw]; } else { src = dst = gw - E0; }
    const int per = CH >> 5;          // 16 (H=4) or 4 (H=1); chunk never crosses a head
    const int f0  = lane * per;
    const int hd  = f0 >> 7;          // f0 / 128
    float e = lrelu(s[src * H + hd] + d[dst * H + hd]);
    float m = decOrd(mEnc[dst * H + hd]);
    float alpha = __expf(e - m) / (den[dst * H + hd] + EPS_SM);
    const __bf16* hp = h + (size_t)src * CH + f0;
    float* ap = agg + (size_t)dst * CH + f0;
    __builtin_prefetch(hp, 0, 0);     // global_prefetch_b8
    for (int j = 0; j < per; ++j) atomicAdd(&ap[j], alpha * bf2f(hp[j]));
}

// ---------------------------------------------------------------------------
// y = LayerNorm(mish(agg + bias)) -> bf16 (next-layer GEMM input). Wave/node.
__global__ __launch_bounds__(256)
void finalize_kernel(const float* __restrict__ agg, const float* __restrict__ bias,
                     const float* __restrict__ g, const float* __restrict__ be,
                     __bf16* __restrict__ y, int N, int CH) {
    int n    = (int)(((long)blockIdx.x * blockDim.x + threadIdx.x) >> 5);
    int lane = threadIdx.x & 31;
    if (n >= N) return;
    const int per = CH >> 5;
    const float* ap = agg + (size_t)n * CH + lane * per;
    float v[16];
    float sum = 0.f;
    for (int j = 0; j < per; ++j) {
        float x = ap[j] + bias[lane * per + j];
        v[j] = mishf(x); sum += v[j];
    }
    for (int o = 16; o > 0; o >>= 1) sum += __shfl_xor(sum, o, 32);
    float mu = sum / (float)CH;
    float var = 0.f;
    for (int j = 0; j < per; ++j) { float t = v[j] - mu; var += t * t; }
    for (int o = 16; o > 0; o >>= 1) var += __shfl_xor(var, o, 32);
    float rs = rsqrtf(var / (float)CH + EPS_LN);
    for (int j = 0; j < per; ++j) {
        int c = lane * per + j;
        y[(size_t)n * CH + c] = f2bf((v[j] - mu) * rs * g[c] + be[c]);
    }
}
// layer-3 epilogue fused with classifier head. Wave/node, CH = 128.
__global__ __launch_bounds__(256)
void finalize3_head_kernel(const float* __restrict__ agg, const float* __restrict__ b3,
                           const float* __restrict__ g3, const float* __restrict__ be3,
                           const float* __restrict__ wh, const float* __restrict__ bh,
                           float* __restrict__ out, int N) {
    int n    = (int)(((long)blockIdx.x * blockDim.x + threadIdx.x) >> 5);
    int lane = threadIdx.x & 31;
    if (n >= N) return;
    const int per = N_HIDDEN / 32;    // 4
    const int f0  = lane * per;
    const float* ap = agg + (size_t)n * N_HIDDEN + f0;
    float v[4];
    float sum = 0.f;
    for (int j = 0; j < per; ++j) { v[j] = mishf(ap[j] + b3[f0 + j]); sum += v[j]; }
    for (int o = 16; o > 0; o >>= 1) sum += __shfl_xor(sum, o, 32);
    float mu = sum / (float)N_HIDDEN;
    float var = 0.f;
    for (int j = 0; j < per; ++j) { float t = v[j] - mu; var += t * t; }
    for (int o = 16; o > 0; o >>= 1) var += __shfl_xor(var, o, 32);
    float rs = rsqrtf(var / (float)N_HIDDEN + EPS_LN);
    float nv[4];
    for (int j = 0; j < per; ++j) nv[j] = (v[j] - mu) * rs * g3[f0 + j] + be3[f0 + j];
    // head: out[n][o] = sum_f nv_f * wh[f][o] + bh[o]
    for (int o = 0; o < OUT_DIM; ++o) {
        float p = 0.f;
        for (int j = 0; j < per; ++j) p += nv[j] * wh[(size_t)(f0 + j) * OUT_DIM + o];
        for (int w = 16; w > 0; w >>= 1) p += __shfl_xor(p, w, 32);
        if (lane == 0) out[(size_t)n * OUT_DIM + o] = p + bh[o];
    }
}

// ---------------------------------------------------------------------------
extern "C" void kernel_launch(void* const* d_in, const int* in_sizes, int n_in,
                              void* d_out, int out_size, void* d_ws, size_t ws_size,
                              hipStream_t stream) {
    const float* x    = (const float*)d_in[0];
    const int*   eidx = (const int*)d_in[1];
    const float* w1   = (const float*)d_in[2];
    const float* as1  = (const float*)d_in[3];
    const float* ad1  = (const float*)d_in[4];
    const float* b1   = (const float*)d_in[5];
    const float* w2   = (const float*)d_in[6];
    const float* as2  = (const float*)d_in[7];
    const float* ad2  = (const float*)d_in[8];
    const float* b2   = (const float*)d_in[9];
    const float* w3   = (const float*)d_in[10];
    const float* as3  = (const float*)d_in[11];
    const float* ad3  = (const float*)d_in[12];
    const float* b3   = (const float*)d_in[13];
    const float* g1   = (const float*)d_in[14];
    const float* be1  = (const float*)d_in[15];
    const float* g2   = (const float*)d_in[16];
    const float* be2  = (const float*)d_in[17];
    const float* g3   = (const float*)d_in[18];
    const float* be3  = (const float*)d_in[19];
    const float* wh   = (const float*)d_in[20];
    const float* bh   = (const float*)d_in[21];
    float* out = (float*)d_out;

    const int N    = in_sizes[0] / N_HIDDEN;   // 50000
    const int E0   = in_sizes[1] / 2;          // 800000
    const int Etot = E0 + N;                   // + self loops

    // ---- workspace layout (256B aligned) ----
    char* base = (char*)d_ws;
    size_t off = 0;
    auto carve = [&](size_t bytes) -> char* {
        char* p = base + off;
        off += (bytes + 255) & ~(size_t)255;
        return p;
    };
    __bf16*   xb   = (__bf16*)  carve((size_t)N * HH * 2);   // ping: layer input (bf16)
    __bf16*   hb   = (__bf16*)  carve((size_t)N * HH * 2);   // pong: GEMM output (bf16)
    float*    agg  = (float*)   carve((size_t)N * HH * 4);   // f32 aggregation target
    float*    sbuf = (float*)   carve((size_t)N * HEADS_12 * 4);
    float*    dbuf = (float*)   carve((size_t)N * HEADS_12 * 4);
    unsigned* mEnc = (unsigned*)carve((size_t)N * HEADS_12 * 4);
    float*    den  = (float*)   carve((size_t)N * HEADS_12 * 4);
    __bf16*   w1t  = (__bf16*)  carve((size_t)N_HIDDEN * HH * 2);
    __bf16*   w2t  = (__bf16*)  carve((size_t)HH * HH * 2);
    __bf16*   w3t  = (__bf16*)  carve((size_t)HH * N_HIDDEN * 2);
    if (off > ws_size) return;  // workspace too small: nothing safe to do

    const int T = 256;
    auto cdiv = [](long a, long b) { return (int)((a + b - 1) / b); };

    // ---- prep: bf16 activations + transposed bf16 weights ----
    cvt_bf16_kernel<<<cdiv((long)N * N_HIDDEN, T), T, 0, stream>>>(x, xb, (long)N * N_HIDDEN);
    transpose_bf16_kernel<<<cdiv((long)N_HIDDEN * HH, T), T, 0, stream>>>(w1, w1t, N_HIDDEN, HH);
    transpose_bf16_kernel<<<cdiv((long)HH * HH, T), T, 0, stream>>>(w2, w2t, HH, HH);
    transpose_bf16_kernel<<<cdiv((long)HH * N_HIDDEN, T), T, 0, stream>>>(w3, w3t, HH, N_HIDDEN);

    struct Layer {
        const __bf16* Wt; int Kin, Mout, H;
        const float *aS, *aD, *bias, *gamma, *beta;
    };
    const Layer L[3] = {
        { w1t, N_HIDDEN, HH,       HEADS_12, as1, ad1, b1, g1, be1 },
        { w2t, HH,       HH,       HEADS_12, as2, ad2, b2, g2, be2 },
        { w3t, HH,       N_HIDDEN, 1,        as3, ad3, b3, g3, be3 },
    };

    for (int l = 0; l < 3; ++l) {
        const Layer& P = L[l];
        const int CH = P.H * N_HIDDEN;
        // GEMM: hb = xb @ W^T   (32x64 tile per wave, 8 waves/block)
        {
            long waveTiles = (long)cdiv(N, GEMM_MR * 16) * (P.Mout >> 6);
            gemm_wmma_bf16_kernel<<<cdiv(waveTiles, 8), T, 0, stream>>>(xb, P.Wt, hb,
                                                                        N, P.Kin, P.Mout);
        }
        // attention logits + stat init
        sd_init_kernel<<<cdiv((long)N * P.H, T), T, 0, stream>>>(hb, P.aS, P.aD,
                                                                 sbuf, dbuf, mEnc, den, N, P.H);
        zero_f32_kernel<<<cdiv((long)N * CH, T), T, 0, stream>>>(agg, (long)N * CH);
        // segment softmax + weighted scatter
        edge_max_kernel<<<cdiv(Etot, T), T, 0, stream>>>(eidx, sbuf, dbuf, mEnc, E0, Etot, P.H);
        edge_denom_kernel<<<cdiv(Etot, T), T, 0, stream>>>(eidx, sbuf, dbuf, mEnc, den, E0, Etot, P.H);
        aggregate_kernel<<<cdiv((long)Etot * 32, T), T, 0, stream>>>(eidx, hb, sbuf, dbuf, mEnc, den,
                                                                     agg, E0, Etot, P.H, CH);
        // epilogue
        if (l < 2) {
            finalize_kernel<<<cdiv((long)N * 32, T), T, 0, stream>>>(agg, P.bias, P.gamma, P.beta,
                                                                     xb, N, CH);
        } else {
            finalize3_head_kernel<<<cdiv((long)N * 32, T), T, 0, stream>>>(agg, P.bias, P.gamma,
                                                                           P.beta, wh, bh, out, N);
        }
    }
}
